// LinearAttention_40570261078614
// MI455X (gfx1250) — matrix operand order repaired and checked by reading, not compiled, not run on hardware
//
#include <hip/hip_runtime.h>
#include <math.h>

// ---------------- problem constants ----------------
#define BATCH 8
#define TLEN  2048
#define DMODEL 512
#define NTOK  (BATCH * TLEN)      // 16384
#define CH    64                  // chunk length
#define NCH   (TLEN / CH)         // 32
#define ESL   32                  // state-column slice per workgroup
#define NESL  (DMODEL / ESL)      // 16

typedef __bf16 bh;
typedef __attribute__((ext_vector_type(16))) bh    v16bf;
typedef __attribute__((ext_vector_type(8)))  bh    v8bf;
typedef __attribute__((ext_vector_type(8)))  float v8f;
typedef __attribute__((ext_vector_type(4))) unsigned int u32x4;
typedef __attribute__((ext_vector_type(8))) int i32x8;
typedef __attribute__((ext_vector_type(4))) int i32x4;

// ---------------- WMMA wrapper (probe-confirmed builtin) ----------------
__device__ __forceinline__ v8f wmma_bf16(v16bf a, v16bf b, v8f c) {
  // D = A(16x32 bf16) * B(32x16 bf16) + C(16x16 f32)
  return __builtin_amdgcn_wmma_f32_16x16x32_bf16(false, a, false, b,
                                                 (short)0, c, false, false);
}

__device__ __forceinline__ v8f vzero8() {
  v8f z;
#pragma unroll
  for (int i = 0; i < 8; ++i) z[i] = 0.f;
  return z;
}

// ---------------- fragment loaders (ISA 7.12.2 layouts, wave32) ----------------
// A fragment, 16x32 tile, row-major bf16 memory (K contiguous).
__device__ __forceinline__ v16bf load_a_rm(const bh* base, int ld, int lane) {
  int row = lane & 15, kb = (lane >> 4) * 8;
  const bh* p = base + row * ld + kb;
  v8bf lo = *(const v8bf*)(p);
  v8bf hi = *(const v8bf*)(p + 16);
  v16bf r;
#pragma unroll
  for (int i = 0; i < 8; ++i) { r[i] = lo[i]; r[i + 8] = hi[i]; }
  return r;
}

// A fragment from row-major f32 memory, converted to bf16 on the fly.
__device__ __forceinline__ v16bf load_a_rm_f32(const float* base, int ld, int lane) {
  int row = lane & 15, kb = (lane >> 4) * 8;
  const float* p = base + row * ld + kb;
  v16bf r;
#pragma unroll
  for (int i = 0; i < 8; ++i) { r[i] = (bh)p[i]; r[i + 8] = (bh)p[16 + i]; }
  return r;
}

// B fragment, 32x16 tile, K-contiguous memory: element (k,n) at base[n*ld + k].
__device__ __forceinline__ v16bf load_b_km(const bh* base, int ld, int lane) {
  int n = lane & 15, kb = (lane >> 4) * 8;
  const bh* p = base + n * ld + kb;
  v8bf lo = *(const v8bf*)(p);
  v8bf hi = *(const v8bf*)(p + 16);
  v16bf r;
#pragma unroll
  for (int i = 0; i < 8; ++i) { r[i] = lo[i]; r[i + 8] = hi[i]; }
  return r;
}

// B fragment, 32x16 tile, N-contiguous (row-major [K][N]) memory: strided per lane.
__device__ __forceinline__ v16bf load_b_nm(const bh* base, int ld, int lane) {
  int n = lane & 15, kb = (lane >> 4) * 8;
  v16bf r;
#pragma unroll
  for (int i = 0; i < 16; ++i) r[i] = base[(kb + i + (i >= 8 ? 8 : 0)) * ld + n];
  return r;
}

// ---------------- kernel 0: weights -> bf16, transposed (WT[m][n][k] = W[k][n]) ----
__global__ __launch_bounds__(256) void prep_weights_kernel(
    const float* __restrict__ Wq, const float* __restrict__ Wk,
    const float* __restrict__ Wv, const float* __restrict__ Wp,
    bh* __restrict__ WT) {
  int idx = blockIdx.x * 256 + threadIdx.x;       // [4][512][512]
  int m = idx >> 18;
  int r = idx & ((1 << 18) - 1);
  int n = r >> 9, k = r & 511;
  const float* W = (m == 0) ? Wq : (m == 1) ? Wk : (m == 2) ? Wv : Wp;
  WT[idx] = (bh)W[k * DMODEL + n];
}

// ---------------- kernel 1: q=phi(xWq), k=phi(xWk), v=xWv  (+ kT scatter) -------
// A fragment (x, converted once) is shared across the three weight matrices.
__global__ __launch_bounds__(256) void qkv_proj_kernel(
    const float* __restrict__ x, const bh* __restrict__ WT,
    bh* __restrict__ qb, bh* __restrict__ kb, bh* __restrict__ vb,
    bh* __restrict__ kT) {
  const int rb = blockIdx.x >> 3;       // row block (64 rows)
  const int cb = blockIdx.x & 7;        // col block (64 cols)
  const int lane = threadIdx.x & 31;
  const int w = threadIdx.x >> 5;

  bh* dsts[3] = {qb, kb, vb};
#pragma unroll
  for (int ti = 0; ti < 2; ++ti) {
    int t = w + ti * 8;                 // 0..15
    int mt = t >> 2, nt = t & 3;
    int r0 = rb * 64 + mt * 16, c0 = cb * 64 + nt * 16;
    v8f acc[3];
#pragma unroll
    for (int m = 0; m < 3; ++m) acc[m] = vzero8();

    for (int kk = 0; kk < DMODEL; kk += 32) {
      v16bf a = load_a_rm_f32(x + (size_t)r0 * DMODEL + kk, DMODEL, lane);
#pragma unroll
      for (int m = 0; m < 3; ++m) {
        const bh* Wt = WT + (size_t)m * DMODEL * DMODEL;   // [n][k], k contiguous
        v16bf bfr = load_b_km(Wt + (size_t)c0 * DMODEL + kk, DMODEL, lane);
        acc[m] = wmma_bf16(a, bfr, acc[m]);
      }
    }
    int M = 8 * (lane >> 4), N = lane & 15;
#pragma unroll
    for (int m = 0; m < 3; ++m) {
      bh* dst = dsts[m];
#pragma unroll
      for (int r = 0; r < 8; ++r) {
        int row = r0 + M + r, col = c0 + N;
        float u = acc[m][r];
        if (m < 2) u = (u > 0.f) ? (u + 1.f) : __expf(u);  // phi = elu+1
        dst[(size_t)row * DMODEL + col] = (bh)u;
        if (m == 1) {                   // transposed copy of k: [B][D][T]
          int bidx = row >> 11, tloc = row & (TLEN - 1);
          kT[((size_t)bidx * DMODEL + col) * TLEN + tloc] = (bh)u;
        }
      }
    }
  }
}

// ---------------- kernel 2: chunked linear-attention scan -----------------------
// grid: BATCH * NESL workgroups; each owns (batch b, state columns e0..e0+31).
__global__ __launch_bounds__(256) void chunk_scan_kernel(
    const bh* __restrict__ qg, const bh* __restrict__ kg,
    const bh* __restrict__ kTg, const bh* __restrict__ vg,
    bh* __restrict__ yg) {
  __shared__ bh    Sb[DMODEL][ESL];   // 32 KB  bf16 mirror of S_prev slice
  __shared__ float As[CH][CH];        // 16 KB  A = Q K^T (full D contraction)
  __shared__ bh    Asb[CH][CH];       // 8 KB   causal-masked bf16 A
  __shared__ bh    Vs[CH][ESL];       // 4 KB   V chunk slice (TDM destination)
  __shared__ float zsh[DMODEL];       // 2 KB   running z
  __shared__ float den[CH];
  __shared__ float dpart[CH][4];

  const int b = blockIdx.x >> 4;
  const int e0 = (blockIdx.x & 15) * ESL;
  const int tid = threadIdx.x;
  const int lane = tid & 31;
  const int w = tid >> 5;

  const bh* qb = qg + (size_t)b * TLEN * DMODEL;
  const bh* kb = kg + (size_t)b * TLEN * DMODEL;
  const bh* vb = vg + (size_t)b * TLEN * DMODEL;
  const bh* kTb = kTg + (size_t)b * DMODEL * TLEN;
  bh* yb = yg + (size_t)b * TLEN * DMODEL;

  // Wave w owns S rows [64w, 64w+64) as 4x2 WMMA accumulator tiles (f32 master).
  v8f sacc[4][2];
#pragma unroll
  for (int i = 0; i < 4; ++i)
#pragma unroll
    for (int j = 0; j < 2; ++j) sacc[i][j] = vzero8();

  for (int i = tid; i < DMODEL * ESL; i += 256) (&Sb[0][0])[i] = (bh)0.f;
  for (int i = tid; i < DMODEL; i += 256) zsh[i] = 0.f;
  __syncthreads();

  for (int c = 0; c < NCH; ++c) {
    const int t0 = c * CH;

    // --- P1: stage V slice via the Tensor Data Mover (async, TENSORcnt) ---
#if __has_builtin(__builtin_amdgcn_tensor_load_to_lds)
    if (w == 0) {
      // D# group 0: count=1, lds_addr[63:32], global_addr[120:64], type=2
      unsigned long long ga =
          (unsigned long long)(uintptr_t)(vb + (size_t)t0 * DMODEL + e0);
      unsigned int lds_off = (unsigned int)(uintptr_t)(&Vs[0][0]);
      u32x4 g0;
      g0[0] = 1u;                                   // count=1, no gather
      g0[1] = lds_off;                              // lds_addr
      g0[2] = (unsigned int)(ga & 0xffffffffu);     // global_addr[31:0]
      g0[3] = (unsigned int)((ga >> 32) & 0x01ffffffu) | (2u << 30); // type=2
      // D# group 1: data_size=2B, tensor_dim=(32,64), tile_dim=(32,64),
      //             tensor_dim0_stride=512 elements
      i32x8 g1;
      g1[0] = 1 << 16;            // data_size=1 (2 bytes), mask=0
      g1[1] = ESL << 16;          // tensor_dim0[15:0] in bits 63:48
      g1[2] = CH << 16;           // tensor_dim0 hi=0 | tensor_dim1[15:0]
      g1[3] = ESL << 16;          // tensor_dim1 hi=0 | tile_dim0
      g1[4] = CH;                 // tile_dim1 | tile_dim2=0
      g1[5] = DMODEL;             // tensor_dim0_stride[31:0]
      g1[6] = 0;                  // stride0 hi | stride1 lo
      g1[7] = 0;                  // stride1 hi
      i32x4 gz4;
      gz4[0] = 0; gz4[1] = 0; gz4[2] = 0; gz4[3] = 0;  // 2D: groups 2/3 unused
      i32x8 gz8;
#pragma unroll
      for (int i = 0; i < 8; ++i) gz8[i] = 0;
      // 6-arg variant (clang-23 / therock-10.0 lane)
      __builtin_amdgcn_tensor_load_to_lds(g0, g1, gz4, gz4, gz8, 0);
    }
#else
    for (int i = tid; i < CH * ESL; i += 256) {
      int t = i / ESL, e = i % ESL;
      Vs[t][e] = vb[(size_t)(t0 + t) * DMODEL + e0 + e];
    }
#endif

    // dvec[t] = q_t . z_prev (4 threads per row) — overlaps with the DMA
    {
      int t = tid >> 2, part = tid & 3;
      const bh* qrow = qb + (size_t)(t0 + t) * DMODEL + part * 128;
      const float* zp = zsh + part * 128;
      float s = 0.f;
#pragma unroll 8
      for (int j = 0; j < 128; ++j) s += (float)qrow[j] * zp[j];
      dpart[t][part] = s;
    }
    if (c + 1 < NCH)  // prefetch next chunk's Q (global_prefetch_b8)
      __builtin_prefetch(qb + (size_t)(t0 + CH) * DMODEL + lane * 64, 0, 1);

    // --- P2: As = Qc @ Kc^T  (B operand K-contiguous from k rows)
#pragma unroll
    for (int ti = 0; ti < 2; ++ti) {
      int t = w * 2 + ti;
      int mt = t >> 2, nt = t & 3;
      v8f acc = vzero8();
      for (int kk = 0; kk < DMODEL; kk += 32) {
        v16bf a = load_a_rm(qb + (size_t)(t0 + mt * 16) * DMODEL + kk, DMODEL, lane);
        v16bf bfr = load_b_km(kb + (size_t)(t0 + nt * 16) * DMODEL + kk, DMODEL, lane);
        acc = wmma_bf16(a, bfr, acc);
      }
      int M = 8 * (lane >> 4), N = lane & 15;
#pragma unroll
      for (int r = 0; r < 8; ++r) As[mt * 16 + M + r][nt * 16 + N] = acc[r];
    }
#if __has_builtin(__builtin_amdgcn_tensor_load_to_lds)
    if (w == 0) __builtin_amdgcn_s_wait_tensorcnt(0);  // Vs resident before barrier
#endif
    __syncthreads();

    // --- P3: den (prefix over As row), masked bf16 As, z update
    if (tid < CH) {
      int t = tid;
      float s = 1e-6f + dpart[t][0] + dpart[t][1] + dpart[t][2] + dpart[t][3];
      for (int tp = 0; tp <= t; ++tp) s += As[t][tp];
      den[t] = s;
    }
    for (int i = tid; i < CH * CH; i += 256) {
      int t = i >> 6, tp = i & 63;
      Asb[t][tp] = (tp <= t) ? (bh)As[t][tp] : (bh)0.f;
    }
    for (int d = tid; d < DMODEL; d += 256) {
      const bh* kr = kTb + (size_t)d * TLEN + t0;
      float s = 0.f;
#pragma unroll 8
      for (int j = 0; j < CH; ++j) s += (float)kr[j];
      zsh[d] += s;
    }
    __syncthreads();

    // --- P4: num = Qc @ S_prev  +  tril(A) @ Vs ; y = num / den
    {
      int mt = w >> 1, et = w & 1;
      v8f acc = vzero8();
      for (int kk = 0; kk < DMODEL; kk += 32) {
        v16bf a = load_a_rm(qb + (size_t)(t0 + mt * 16) * DMODEL + kk, DMODEL, lane);
        v16bf bfr = load_b_nm(&Sb[kk][et * 16], ESL, lane);
        acc = wmma_bf16(a, bfr, acc);
      }
#pragma unroll
      for (int kk = 0; kk < CH; kk += 32) {
        v16bf a = load_a_rm(&Asb[mt * 16][kk], CH, lane);
        v16bf bfr = load_b_nm(&Vs[kk][et * 16], ESL, lane);
        acc = wmma_bf16(a, bfr, acc);
      }
      int M = 8 * (lane >> 4), N = lane & 15;
#pragma unroll
      for (int r = 0; r < 8; ++r) {
        int t = mt * 16 + M + r;
        float val = acc[r] / den[t];
        yb[(size_t)(t0 + t) * DMODEL + e0 + et * 16 + N] = (bh)val;
      }
    }
    __syncthreads();  // everyone done reading Sb before the update rewrites it

    // --- P5: S += Kc^T @ Vs (f32 accumulators), refresh bf16 mirror
#pragma unroll
    for (int i = 0; i < 4; ++i) {
      int dt = w * 4 + i;
#pragma unroll
      for (int j = 0; j < 2; ++j) {
        v8f acc = sacc[i][j];
#pragma unroll
        for (int kk = 0; kk < CH; kk += 32) {
          v16bf a = load_a_rm(kTb + (size_t)(dt * 16) * TLEN + t0 + kk, TLEN, lane);
          v16bf bfr = load_b_nm(&Vs[kk][j * 16], ESL, lane);
          acc = wmma_bf16(a, bfr, acc);
        }
        sacc[i][j] = acc;
        int M = 8 * (lane >> 4), N = lane & 15;
#pragma unroll
        for (int r = 0; r < 8; ++r)
          Sb[dt * 16 + M + r][j * 16 + N] = (bh)acc[r];
      }
    }
    __syncthreads();  // Sb / zsh consistent, Vs free for next chunk's TDM
  }
}

// ---------------- kernel 3: out = y @ Wp (f32 output) ---------------------------
__global__ __launch_bounds__(256) void out_proj_kernel(
    const bh* __restrict__ y, const bh* __restrict__ WpT,
    float* __restrict__ out) {
  const int rb = blockIdx.x >> 3;
  const int cb = blockIdx.x & 7;
  const int lane = threadIdx.x & 31;
  const int w = threadIdx.x >> 5;
#pragma unroll
  for (int ti = 0; ti < 2; ++ti) {
    int t = w + ti * 8;
    int mt = t >> 2, nt = t & 3;
    int r0 = rb * 64 + mt * 16, c0 = cb * 64 + nt * 16;
    v8f acc = vzero8();
    for (int kk = 0; kk < DMODEL; kk += 32) {
      v16bf a = load_a_rm(y + (size_t)r0 * DMODEL + kk, DMODEL, lane);
      v16bf bfr = load_b_km(WpT + (size_t)c0 * DMODEL + kk, DMODEL, lane);
      acc = wmma_bf16(a, bfr, acc);
    }
    int M = 8 * (lane >> 4), N = lane & 15;
#pragma unroll
    for (int r = 0; r < 8; ++r)
      out[(size_t)(r0 + M + r) * DMODEL + c0 + N] = acc[r];
  }
}

// ---------------- launcher -------------------------------------------------------
extern "C" void kernel_launch(void* const* d_in, const int* in_sizes, int n_in,
                              void* d_out, int out_size, void* d_ws, size_t ws_size,
                              hipStream_t stream) {
  (void)in_sizes; (void)n_in; (void)out_size; (void)ws_size;
  const float* x  = (const float*)d_in[0];
  const float* Wq = (const float*)d_in[1];
  const float* Wk = (const float*)d_in[2];
  const float* Wv = (const float*)d_in[3];
  const float* Wp = (const float*)d_in[4];
  float* out = (float*)d_out;

  // workspace layout (bf16): 4 transposed weights + q,k,v,kT,y  (~82 MB)
  bh* WT   = (bh*)d_ws;                              // [4][512][512]
  bh* qbuf = WT + (size_t)4 * DMODEL * DMODEL;
  bh* kbuf = qbuf + (size_t)NTOK * DMODEL;
  bh* vbuf = kbuf + (size_t)NTOK * DMODEL;
  bh* kTbf = vbuf + (size_t)NTOK * DMODEL;
  bh* ybuf = kTbf + (size_t)NTOK * DMODEL;

  prep_weights_kernel<<<(4 * DMODEL * DMODEL) / 256, 256, 0, stream>>>(
      Wq, Wk, Wv, Wp, WT);
  qkv_proj_kernel<<<(NTOK / 64) * (DMODEL / 64), 256, 0, stream>>>(
      x, WT, qbuf, kbuf, vbuf, kTbf);
  chunk_scan_kernel<<<BATCH * NESL, 256, 0, stream>>>(
      qbuf, kbuf, kTbf, vbuf, ybuf);
  out_proj_kernel<<<(NTOK / 64) * (DMODEL / 64), 256, 0, stream>>>(
      ybuf, WT + (size_t)3 * DMODEL * DMODEL, out);
}
